// AtomFormerBlock_79087527789125
// MI455X (gfx1250) — matrix operand order repaired and checked by reading, not compiled, not run on hardware
//
#include <hip/hip_runtime.h>

// ---------------------------------------------------------------------------
// CDNA5 (gfx1250) wave32 WMMA types
// ---------------------------------------------------------------------------
typedef __bf16        v16bf __attribute__((ext_vector_type(16)));
typedef float         v8f   __attribute__((ext_vector_type(8)));

struct bfrag32 { uint4 lo, hi; };  // 32 bytes = one v16bf fragment

__device__ __forceinline__ v8f wmma_bf16(v16bf a, v16bf b, v8f c) {
  return __builtin_amdgcn_wmma_f32_16x16x32_bf16(false, a, false, b, (short)0, c,
                                                 false, false);
}

// ---------------------------------------------------------------------------
// gfx1250 async copy: one 16B global->LDS transfer per lane, ASYNCcnt-tracked.
// (CDNA5 ISA; inline asm per CDNA5_HIP.md guidance)
// ---------------------------------------------------------------------------
__device__ __forceinline__ void async_b128(void* lds, const void* gptr) {
  asm volatile("global_load_async_to_lds_b128 %0, %1, off"
               :: "v"((unsigned int)(unsigned long long)lds), "v"(gptr)
               : "memory");
}
__device__ __forceinline__ void wait_async0() {
  asm volatile("s_wait_asynccnt 0x0" ::: "memory");
}

// A-fragment 16x32 bf16 (MxK).  Lane r = lane&15 is the row, lanes>=16 take
// K offset +8.  Per-lane data = two contiguous 16B runs.
__device__ __forceinline__ v16bf load_frag_a(const __bf16* base, int stride) {
  const int lane = threadIdx.x & 31;
  const __bf16* p = base + (size_t)(lane & 15) * stride + ((lane >> 4) << 3);
  bfrag32 u;
  u.lo = *(const uint4*)(p);
  u.hi = *(const uint4*)(p + 16);
  return __builtin_bit_cast(v16bf, u);
}

// B-fragment 32x16 bf16 (KxN).  Lane c = lane&15 is the column; lanes>=16
// take K offset +8.  Per-lane data = two contiguous 16B runs.
__device__ __forceinline__ v16bf load_frag_b(const __bf16* base, int stride) {
  const int lane = threadIdx.x & 31;
  const __bf16* p = base + (size_t)(lane & 15) * stride + ((lane >> 4) << 3);
  bfrag32 u;
  u.lo = *(const uint4*)(p);
  u.hi = *(const uint4*)(p + 16);
  return __builtin_bit_cast(v16bf, u);
}

__device__ __forceinline__ unsigned int pack2bf(float a, float b) {
  const unsigned int ua = __builtin_bit_cast(unsigned short, (__bf16)a);
  const unsigned int ub = __builtin_bit_cast(unsigned short, (__bf16)b);
  return ua | (ub << 16);
}

// ---------------------------------------------------------------------------
// GEMM core (double-buffered, async staging):
// workgroup (4 waves) computes a 32-row x 256-col tile of A(bf16 row-major)
// x WT (bf16 pre-transposed col-major WT[col][K]).
// aT: 2 x [32][32] bf16.  wT: 2 x [256][32] bf16 (K-pairs contiguous).
// Tile k+1 is async-prefetched into the alternate buffer while computing k.
// ---------------------------------------------------------------------------
__device__ __forceinline__ void stage_tiles(const __bf16* __restrict__ A, int lda,
                                            const __bf16* __restrict__ WT, int ldk,
                                            int row0, int col0, int k0,
                                            __bf16* aT, __bf16* wT) {
  const int tid = threadIdx.x;
  {  // A tile: 32x32 bf16 = 128 x 16B, one per thread
    const int r = tid >> 2, c8 = (tid & 3) * 8;
    async_b128(aT + r * 32 + c8, A + (size_t)(row0 + r) * lda + k0 + c8);
  }
#pragma unroll
  for (int t = 0; t < 8; ++t) {  // W tile: 256x32 bf16 = 1024 x 16B
    const int idx = t * 128 + tid;
    const int col = idx >> 2, c8 = (idx & 3) * 8;
    async_b128(wT + col * 32 + c8, WT + (size_t)(col0 + col) * ldk + k0 + c8);
  }
}

__device__ __forceinline__ void compute_step(const __bf16* aT, const __bf16* wT,
                                             int wave, v8f acc[2][4]) {
  const v16bf a0 = load_frag_a(aT, 32);
  const v16bf a1 = load_frag_a(aT + 16 * 32, 32);
#pragma unroll
  for (int j = 0; j < 4; ++j) {
    const v16bf b = load_frag_b(wT + (size_t)(wave * 64 + j * 16) * 32, 32);
    acc[0][j] = wmma_bf16(a0, b, acc[0][j]);
    acc[1][j] = wmma_bf16(a1, b, acc[1][j]);
  }
}

__device__ __forceinline__ void gemm32x256(const __bf16* __restrict__ A, int lda,
                                           const __bf16* __restrict__ WT, int ldk,
                                           int row0, int col0, int K,
                                           __bf16* aT, __bf16* wT, v8f acc[2][4]) {
  const int wave = threadIdx.x >> 5;
  stage_tiles(A, lda, WT, ldk, row0, col0, 0, aT, wT);
  wait_async0();
  __syncthreads();
  int cur = 0;
  for (int k0 = 32; k0 < K; k0 += 32) {
    const int nxt = cur ^ 1;
    stage_tiles(A, lda, WT, ldk, row0, col0, k0, aT + nxt * 1024, wT + nxt * 8192);
    compute_step(aT + cur * 1024, wT + cur * 8192, wave, acc);
    wait_async0();
    __syncthreads();
    cur = nxt;
  }
  compute_step(aT + cur * 1024, wT + cur * 8192, wave, acc);
}

#define ZACC(acc)                                                   \
  {                                                                 \
    const v8f z_ = {0.f, 0.f, 0.f, 0.f, 0.f, 0.f, 0.f, 0.f};        \
    acc[0][0] = z_; acc[0][1] = z_; acc[0][2] = z_; acc[0][3] = z_; \
    acc[1][0] = z_; acc[1][1] = z_; acc[1][2] = z_; acc[1][3] = z_; \
  }

// ---------------------------------------------------------------------------
// Conversion kernels (once per launch; ~7 MB total -> negligible at 23 TB/s).
// ---------------------------------------------------------------------------
__global__ __launch_bounds__(256) void k_wt(const float* __restrict__ W,
                                            __bf16* __restrict__ WT,
                                            int K, int Ncols) {
  const int i = blockIdx.x * 256 + threadIdx.x;  // over K*N/2
  const int kh = K >> 1;
  if (i >= kh * Ncols) return;
  const int col = i / kh;
  const int k2  = (i - col * kh) * 2;
  const float a = W[(size_t)k2 * Ncols + col];
  const float b = W[(size_t)(k2 + 1) * Ncols + col];
  *(unsigned int*)(WT + (size_t)col * K + k2) = pack2bf(a, b);
}

__global__ __launch_bounds__(256) void k_cvt(const float* __restrict__ X,
                                             __bf16* __restrict__ Xb, int total4) {
  const int i = blockIdx.x * 256 + threadIdx.x;
  if (i >= total4) return;
  const float4 v = ((const float4*)X)[i];
  uint2 o;
  o.x = pack2bf(v.x, v.y);
  o.y = pack2bf(v.z, v.w);
  ((uint2*)Xb)[i] = o;
}

// ---------------------------------------------------------------------------
// Kernel 1: fused q/k/v/gate projection.  grid = (256 row tiles, 16 col blks).
// ---------------------------------------------------------------------------
__global__ __launch_bounds__(128) void k_qkvg(
    const __bf16* __restrict__ X, const __bf16* __restrict__ wqT,
    const __bf16* __restrict__ wkT, const __bf16* __restrict__ wvT,
    const __bf16* __restrict__ wgT, const float* __restrict__ bg,
    __bf16* __restrict__ qb, __bf16* __restrict__ kbuf,
    __bf16* __restrict__ vtb, float* __restrict__ gtb) {
  __shared__ __bf16 aT[2][32 * 32];
  __shared__ __bf16 wT[2][256 * 32];
  const int row0 = blockIdx.x * 32;
  const int cb   = blockIdx.y;
  const int wsel = cb >> 2;
  const int col0 = (cb & 3) * 256;
  const __bf16* WT = (wsel == 0) ? wqT : (wsel == 1) ? wkT : (wsel == 2) ? wvT : wgT;
  v8f acc[2][4];
  ZACC(acc);
  gemm32x256(X, 256, WT, 256, row0, col0, 256, aT[0], wT[0], acc);

  const int lane = threadIdx.x & 31, wave = threadIdx.x >> 5;
  const int half = lane >> 4, lcol = lane & 15;
#pragma unroll
  for (int rb = 0; rb < 2; ++rb) {
#pragma unroll
    for (int j = 0; j < 4; ++j) {
#pragma unroll
      for (int e = 0; e < 8; ++e) {
        const int gcol = col0 + wave * 64 + j * 16 + lcol;
        const int grow = row0 + rb * 16 + e + half * 8;
        const int b = grow >> 10, n = grow & 1023;
        const int d = gcol >> 2, h = gcol & 3;  // reshape (.., d, H): h fastest
        const size_t bh = (size_t)(b * 4 + h);
        const float val = acc[rb][j][e];
        if (wsel == 0)
          qb[(bh * 1024 + n) * 256 + d] = (__bf16)val;
        else if (wsel == 1)
          kbuf[(bh * 1024 + n) * 256 + d] = (__bf16)val;
        else if (wsel == 2)
          vtb[(bh * 256 + d) * 1024 + n] = (__bf16)val;
        else {
          const float z = val + bg[gcol];
          gtb[(bh * 1024 + n) * 256 + d] = 1.0f / (1.0f + __expf(-z));
        }
      }
    }
  }
}

// ---------------------------------------------------------------------------
// Kernel 2: flash attention with inline pairwise bias.
// grid = (B*H = 32, N/64 = 16); block = 128 (4 waves x 16 q rows).
// ---------------------------------------------------------------------------
__global__ __launch_bounds__(128) void k_attn(
    const __bf16* __restrict__ qb, const __bf16* __restrict__ kbuf,
    const __bf16* __restrict__ vtb, const float* __restrict__ gtb,
    const float* __restrict__ mm, const unsigned char* __restrict__ msk,
    const float* __restrict__ lbg, const float* __restrict__ lbb,
    const float* __restrict__ wbias, __bf16* __restrict__ attn_out) {
  __shared__ __bf16 kT[32 * 256];    // [key][d]   (K-pairs along d)
  __shared__ __bf16 vT[256 * 32];    // [d][key]   (K-pairs along key)
  __shared__ __bf16 pT[4][16 * 32];  // per-wave P relayout buffer

  const int tid = threadIdx.x, wave = tid >> 5, lane = tid & 31;
  const int half = lane >> 4, lcol = lane & 15;
  const int b = blockIdx.x >> 2, h = blockIdx.x & 3;
  const int qbase = blockIdx.y * 64 + wave * 16;
  const size_t bh = (size_t)(b * 4 + h);

  const __bf16* qB  = qb + bh * 1024 * 256;
  const __bf16* kB  = kbuf + bh * 1024 * 256;
  const __bf16* vtB = vtb + bh * 256 * 1024;

  const float g0 = lbg[0], g1 = lbg[1], g2 = lbg[2], g3 = lbg[3];
  const float o0 = lbb[0], o1 = lbb[1], o2 = lbb[2], o3 = lbb[3];
  const float wb0 = wbias[0 * 4 + h], wb1 = wbias[1 * 4 + h];
  const float wb2 = wbias[2 * 4 + h], wb3 = wbias[3 * 4 + h];

  // Q fragments are key-tile invariant: hoist into registers (8 x 8 VGPR).
  v16bf qfrag[8];
#pragma unroll
  for (int c = 0; c < 8; ++c)
    qfrag[c] = load_frag_a(qB + (size_t)qbase * 256 + c * 32, 256);

  const v8f zero = {0.f, 0.f, 0.f, 0.f, 0.f, 0.f, 0.f, 0.f};
  v8f accO[16];
#pragma unroll
  for (int i = 0; i < 16; ++i) accO[i] = zero;
  float mrun[8], lrun[8];
#pragma unroll
  for (int e = 0; e < 8; ++e) { mrun[e] = -3.0e38f; lrun[e] = 0.f; }

  for (int kt0 = 0; kt0 < 1024; kt0 += 32) {
    // ---- async-stage K tile [32][256] and V^T tile [256][32] -------------
#pragma unroll
    for (int t = 0; t < 8; ++t) {
      const int idx = t * 128 + tid;  // 0..1023
      const int key = idx >> 5, c16 = (idx & 31) * 8;
      async_b128(kT + key * 256 + c16,
                 kB + (size_t)(kt0 + key) * 256 + c16);
    }
#pragma unroll
    for (int t = 0; t < 8; ++t) {
      const int idx = t * 128 + tid;
      const int d = idx >> 2, c8 = (idx & 3) * 8;
      async_b128(vT + d * 32 + c8,
                 vtB + (size_t)d * 1024 + kt0 + c8);
    }
    wait_async0();
    __syncthreads();

    // ---- S = Q K^T : keys 0-15 -> s0acc, keys 16-31 -> s1acc -------------
    v8f s0acc = zero, s1acc = zero;
#pragma unroll
    for (int c = 0; c < 8; ++c) {
      const v16bf bl = load_frag_b(kT + c * 32, 256);
      const v16bf br = load_frag_b(kT + (size_t)16 * 256 + c * 32, 256);
      s0acc = wmma_bf16(qfrag[c], bl, s0acc);
      s1acc = wmma_bf16(qfrag[c], br, s1acc);
    }

    // ---- inline bias (LayerNorm over C=4 + w_bias), mask, scale ----------
    float s0[8], s1[8];
#pragma unroll
    for (int e = 0; e < 8; ++e) {
      const int grow = qbase + e + half * 8;
      const size_t rb = ((size_t)b * 1024 + grow) * 1024;
#pragma unroll
      for (int side = 0; side < 2; ++side) {
        const int key = kt0 + lcol + side * 16;
        const float4 m4 = *(const float4*)(mm + (rb + key) * 4);
        const float mu = 0.25f * (m4.x + m4.y + m4.z + m4.w);
        const float va = 0.25f * (m4.x * m4.x + m4.y * m4.y +
                                  m4.z * m4.z + m4.w * m4.w) - mu * mu;
        const float rs = rsqrtf(va + 1e-5f);
        const float bias = ((m4.x - mu) * rs * g0 + o0) * wb0 +
                           ((m4.y - mu) * rs * g1 + o1) * wb1 +
                           ((m4.z - mu) * rs * g2 + o2) * wb2 +
                           ((m4.w - mu) * rs * g3 + o3) * wb3;
        float s = (side ? s1acc[e] : s0acc[e]) + bias;
        if (!msk[(size_t)b * 1024 + key]) s = -3.0e38f;
        s *= 0.0625f;  // 1/sqrt(D)
        if (side) s1[e] = s; else s0[e] = s;
      }
    }

    // ---- online softmax (row reductions across 16-lane groups) -----------
#pragma unroll
    for (int e = 0; e < 8; ++e) {
      float mx = fmaxf(s0[e], s1[e]);
#pragma unroll
      for (int off = 1; off < 16; off <<= 1) mx = fmaxf(mx, __shfl_xor(mx, off, 16));
      const float mnew = fmaxf(mrun[e], mx);
      const float corr = __expf(mrun[e] - mnew);
      mrun[e] = mnew;
      const float p0 = __expf(s0[e] - mnew);
      const float p1 = __expf(s1[e] - mnew);
      float rsum = p0 + p1;
#pragma unroll
      for (int off = 1; off < 16; off <<= 1) rsum += __shfl_xor(rsum, off, 16);
      lrun[e] = lrun[e] * corr + rsum;
#pragma unroll
      for (int dch = 0; dch < 16; ++dch) accO[dch][e] *= corr;
      // C-layout -> row-major P tile [16][32] in LDS
      pT[wave][(e + half * 8) * 32 + lcol]      = (__bf16)p0;
      pT[wave][(e + half * 8) * 32 + lcol + 16] = (__bf16)p1;
    }
    __syncthreads();

    // ---- O += P V --------------------------------------------------------
    const v16bf pfrag = load_frag_a(&pT[wave][0], 32);
#pragma unroll
    for (int dch = 0; dch < 16; ++dch) {
      const v16bf bv = load_frag_b(vT + (size_t)(dch * 16) * 32, 32);
      accO[dch] = wmma_bf16(pfrag, bv, accO[dch]);
    }
    __syncthreads();
  }

  // ---- epilogue: normalize, gate, scatter bf16 to (B,N,D*H), col=d*4+h ---
#pragma unroll
  for (int dch = 0; dch < 16; ++dch) {
#pragma unroll
    for (int e = 0; e < 8; ++e) {
      const int grow = qbase + e + half * 8;
      const int d = dch * 16 + lcol;
      const float val = accO[dch][e] / lrun[e];
      const float gt = gtb[(bh * 1024 + grow) * 256 + d];
      attn_out[((size_t)b * 1024 + grow) * 1024 + d * 4 + h] = (__bf16)(val * gt);
    }
  }
}

// ---------------------------------------------------------------------------
// Kernel 3: out projection (1024->256) + bias + fused LayerNorm -> bf16.
// ---------------------------------------------------------------------------
__global__ __launch_bounds__(128) void k_oproj_ln(
    const __bf16* __restrict__ AO, const __bf16* __restrict__ woT,
    const float* __restrict__ bo, const float* __restrict__ lng,
    const float* __restrict__ lnb, __bf16* __restrict__ yn) {
  __shared__ __bf16 aT[2][32 * 32];
  __shared__ __bf16 wT[2][256 * 32];
  __shared__ float redS[4][32], redQ[4][32];
  const int row0 = blockIdx.x * 32;
  v8f acc[2][4];
  ZACC(acc);
  gemm32x256(AO, 1024, woT, 1024, row0, 0, 1024, aT[0], wT[0], acc);

  const int lane = threadIdx.x & 31, wave = threadIdx.x >> 5;
  const int half = lane >> 4, lcol = lane & 15;
#pragma unroll
  for (int j = 0; j < 4; ++j) {
    const float bb = bo[wave * 64 + j * 16 + lcol];
#pragma unroll
    for (int rb = 0; rb < 2; ++rb)
#pragma unroll
      for (int e = 0; e < 8; ++e) acc[rb][j][e] += bb;
  }
#pragma unroll
  for (int rb = 0; rb < 2; ++rb) {
#pragma unroll
    for (int e = 0; e < 8; ++e) {
      float s = acc[rb][0][e] + acc[rb][1][e] + acc[rb][2][e] + acc[rb][3][e];
      float q = acc[rb][0][e] * acc[rb][0][e] + acc[rb][1][e] * acc[rb][1][e] +
                acc[rb][2][e] * acc[rb][2][e] + acc[rb][3][e] * acc[rb][3][e];
#pragma unroll
      for (int off = 1; off < 16; off <<= 1) {
        s += __shfl_xor(s, off, 16);
        q += __shfl_xor(q, off, 16);
      }
      if (lcol == 0) {
        redS[wave][rb * 16 + half * 8 + e] = s;
        redQ[wave][rb * 16 + half * 8 + e] = q;
      }
    }
  }
  __syncthreads();
#pragma unroll
  for (int rb = 0; rb < 2; ++rb) {
#pragma unroll
    for (int e = 0; e < 8; ++e) {
      const int row = rb * 16 + half * 8 + e;
      const float S = redS[0][row] + redS[1][row] + redS[2][row] + redS[3][row];
      const float Q = redQ[0][row] + redQ[1][row] + redQ[2][row] + redQ[3][row];
      const float mu = S * (1.0f / 256.0f);
      const float var = Q * (1.0f / 256.0f) - mu * mu;
      const float rs = rsqrtf(var + 1e-5f);
      const size_t grow = (size_t)row0 + row;
#pragma unroll
      for (int j = 0; j < 4; ++j) {
        const int gcol = wave * 64 + j * 16 + lcol;
        yn[grow * 256 + gcol] =
            (__bf16)((acc[rb][j][e] - mu) * rs * lng[gcol] + lnb[gcol]);
      }
    }
  }
}

// ---------------------------------------------------------------------------
// Kernel 4: FFN up (256 -> 1024) + ReLU -> bf16.  grid = (256, 4).
// ---------------------------------------------------------------------------
__global__ __launch_bounds__(128) void k_ffn1(
    const __bf16* __restrict__ yn, const __bf16* __restrict__ w1T,
    const float* __restrict__ b1, __bf16* __restrict__ h1) {
  __shared__ __bf16 aT[2][32 * 32];
  __shared__ __bf16 wT[2][256 * 32];
  const int row0 = blockIdx.x * 32;
  const int col0 = blockIdx.y * 256;
  v8f acc[2][4];
  ZACC(acc);
  gemm32x256(yn, 256, w1T, 256, row0, col0, 256, aT[0], wT[0], acc);

  const int lane = threadIdx.x & 31, wave = threadIdx.x >> 5;
  const int half = lane >> 4, lcol = lane & 15;
#pragma unroll
  for (int j = 0; j < 4; ++j) {
    const int gcol = col0 + wave * 64 + j * 16 + lcol;
    const float bb = b1[gcol];
#pragma unroll
    for (int rb = 0; rb < 2; ++rb)
#pragma unroll
      for (int e = 0; e < 8; ++e) {
        const size_t grow = (size_t)row0 + rb * 16 + e + half * 8;
        h1[grow * 1024 + gcol] = (__bf16)fmaxf(acc[rb][j][e] + bb, 0.0f);
      }
  }
}

// ---------------------------------------------------------------------------
// Kernel 5: FFN down (1024 -> 256) + bias -> fp32 output.  grid = 256.
// ---------------------------------------------------------------------------
__global__ __launch_bounds__(128) void k_ffn2(
    const __bf16* __restrict__ h1, const __bf16* __restrict__ w2T,
    const float* __restrict__ b2, float* __restrict__ out) {
  __shared__ __bf16 aT[2][32 * 32];
  __shared__ __bf16 wT[2][256 * 32];
  const int row0 = blockIdx.x * 32;
  v8f acc[2][4];
  ZACC(acc);
  gemm32x256(h1, 1024, w2T, 1024, row0, 0, 1024, aT[0], wT[0], acc);

  const int lane = threadIdx.x & 31, wave = threadIdx.x >> 5;
  const int half = lane >> 4, lcol = lane & 15;
#pragma unroll
  for (int j = 0; j < 4; ++j) {
    const int gcol = wave * 64 + j * 16 + lcol;
    const float bb = b2[gcol];
#pragma unroll
    for (int rb = 0; rb < 2; ++rb)
#pragma unroll
      for (int e = 0; e < 8; ++e) {
        const size_t grow = (size_t)row0 + rb * 16 + e + half * 8;
        out[grow * 256 + gcol] = acc[rb][j][e] + bb;
      }
  }
}

// ---------------------------------------------------------------------------
// Host launcher
// ---------------------------------------------------------------------------
extern "C" void kernel_launch(void* const* d_in, const int* in_sizes, int n_in,
                              void* d_out, int out_size, void* d_ws, size_t ws_size,
                              hipStream_t stream) {
  (void)in_sizes; (void)n_in; (void)out_size; (void)ws_size;
  const float* mm    = (const float*)d_in[0];   // (B,N,N,C)
  const float* xe    = (const float*)d_in[1];   // (B,N,D)
  const unsigned char* msk = (const unsigned char*)d_in[2];  // (B,N) bool
  const float* lbg   = (const float*)d_in[3];
  const float* lbb   = (const float*)d_in[4];
  const float* wbias = (const float*)d_in[5];
  const float* wq    = (const float*)d_in[6];
  const float* wk    = (const float*)d_in[7];
  const float* wv    = (const float*)d_in[8];
  const float* wg    = (const float*)d_in[9];
  const float* bg    = (const float*)d_in[10];
  const float* wo    = (const float*)d_in[11];
  const float* bo    = (const float*)d_in[12];
  const float* lng   = (const float*)d_in[13];
  const float* lnb   = (const float*)d_in[14];
  const float* w1    = (const float*)d_in[15];
  const float* b1    = (const float*)d_in[16];
  const float* w2    = (const float*)d_in[17];
  const float* b2    = (const float*)d_in[18];
  float* out = (float*)d_out;

  char* ws = (char*)d_ws;
  __bf16* qb   = (__bf16*)(ws + 0);                // 16 MB [B][H][N][D] bf16
  __bf16* kbuf = (__bf16*)(ws + (16ull  << 20));   // 16 MB [B][H][N][D] bf16
  __bf16* vtb  = (__bf16*)(ws + (32ull  << 20));   // 16 MB [B][H][D][N] bf16
  float*  gtb  = (float*) (ws + (48ull  << 20));   // 32 MB [B][H][N][D] f32
  __bf16* aob  = (__bf16*)(ws + (80ull  << 20));   // 16 MB (B,N,1024) bf16
  __bf16* ynb  = (__bf16*)(ws + (96ull  << 20));   //  4 MB (8192,256) bf16
  __bf16* h1b  = (__bf16*)(ws + (100ull << 20));   // 16 MB (8192,1024) bf16
  __bf16* xeb  = (__bf16*)(ws + (116ull << 20));   //  4 MB (8192,256) bf16
  __bf16* wqT  = (__bf16*)(ws + (120ull << 20));            // 512 KB each
  __bf16* wkT  = (__bf16*)(ws + (120ull << 20) + 1 * 524288);
  __bf16* wvT  = (__bf16*)(ws + (120ull << 20) + 2 * 524288);
  __bf16* wgT  = (__bf16*)(ws + (120ull << 20) + 3 * 524288);
  __bf16* woT  = (__bf16*)(ws + (120ull << 20) + 4 * 524288);
  __bf16* w1T  = (__bf16*)(ws + (120ull << 20) + 5 * 524288);
  __bf16* w2T  = (__bf16*)(ws + (120ull << 20) + 6 * 524288);

  // --- one-time conversions (bf16 activations, pre-transposed bf16 weights)
  k_cvt<<<dim3(2048), 256, 0, stream>>>(xe, xeb, 8192 * 256 / 4);
  k_wt<<<dim3(512), 256, 0, stream>>>(wq, wqT, 256, 1024);
  k_wt<<<dim3(512), 256, 0, stream>>>(wk, wkT, 256, 1024);
  k_wt<<<dim3(512), 256, 0, stream>>>(wv, wvT, 256, 1024);
  k_wt<<<dim3(512), 256, 0, stream>>>(wg, wgT, 256, 1024);
  k_wt<<<dim3(512), 256, 0, stream>>>(wo, woT, 1024, 256);
  k_wt<<<dim3(512), 256, 0, stream>>>(w1, w1T, 256, 1024);
  k_wt<<<dim3(512), 256, 0, stream>>>(w2, w2T, 1024, 256);

  // --- main pipeline
  k_qkvg<<<dim3(256, 16), 128, 0, stream>>>(xeb, wqT, wkT, wvT, wgT, bg,
                                            qb, kbuf, vtb, gtb);
  k_attn<<<dim3(32, 16), 128, 0, stream>>>(qb, kbuf, vtb, gtb, mm, msk,
                                           lbg, lbb, wbias, aob);
  k_oproj_ln<<<dim3(256), 128, 0, stream>>>(aob, woT, bo, lng, lnb, ynb);
  k_ffn1<<<dim3(256, 4), 128, 0, stream>>>(ynb, w1T, b1, h1b);
  k_ffn2<<<dim3(256), 128, 0, stream>>>(h1b, w2T, b2, out);
}
